// MultiHeadAttention_68461778698742
// MI455X (gfx1250) — matrix operand order repaired and checked by reading, not compiled, not run on hardware
//
#include <hip/hip_runtime.h>
#include <hip/hip_bf16.h>

// MHA forward B=2, S=2048, D=2048, H=16, hd=128 on gfx1250 (MI455X).
// - All matmuls via v_wmma_f32_16x16x32_f16 (wave32 WMMA, f32 accumulate)
// - Projections: LDS-staged GEMM double-buffered with global_load_async_to_lds_b128
//   (ASYNCcnt) so DMA overlaps WMMA
// - Attention: flash-style online softmax, register-pipelined K/V fragment prefetch

typedef __attribute__((ext_vector_type(16))) _Float16 v16h;
typedef __attribute__((ext_vector_type(8)))  _Float16 v8h;
typedef __attribute__((ext_vector_type(8)))  float    v8f;

#define BATCH 2
#define SEQ   2048
#define DMODEL 2048
#define NHEAD 16
#define HDIM  128
#define MROWS (BATCH * SEQ)       // 4096

__device__ __forceinline__ v8f wmma_f16(v16h a, v16h b, v8f c) {
  return __builtin_amdgcn_wmma_f32_16x16x32_f16(
      false, a, false, b, (short)0, c, false, false);
}

// A-fragment (16x32 f16, row-major, stride lda halves).
// lane l (m=l&15, g=l>>4): elem e = K=g*16+e (contiguous).
__device__ __forceinline__ v16h load_a_frag(const _Float16* base, int lda, int lane) {
  int m = lane & 15, g = lane >> 4;
  v8h lo = *(const v8h*)(base + (size_t)m * lda + g * 16);
  v8h hi = *(const v8h*)(base + (size_t)m * lda + g * 16 + 8);
  v16h a;
#pragma unroll
  for (int e = 0; e < 8; e++) { a[e] = lo[e]; a[8 + e] = hi[e]; }
  return a;
}

// B-fragment (32x16 f16) from N-major source BT[n*ldb + k].
// lane l (n=l&15, g=l>>4): elem j = B[g*16+j][n] = BT[n][g*16+j] (contiguous).
__device__ __forceinline__ v16h load_bt_frag(const _Float16* baseT, int ldb, int lane) {
  int n = lane & 15, g = lane >> 4;
  return *(const v16h*)(baseT + (size_t)n * ldb + g * 16);
}

// CDNA5 async copy: 16B per lane, global -> LDS, tracked by ASYNCcnt.
__device__ __forceinline__ void async_b128(const _Float16* g, _Float16* lds) {
  unsigned loff = (unsigned)(uintptr_t)lds;   // generic LDS ptr: low 32 bits = LDS offset
  asm volatile("global_load_async_to_lds_b128 %0, %1, off"
               :: "v"(loff), "v"(g) : "memory");
}

// ---------------- conversion kernels ----------------

__global__ void f32_to_f16_kernel(const float* __restrict__ in,
                                  _Float16* __restrict__ out, int n) {
  int i = blockIdx.x * blockDim.x + threadIdx.x;
  if (i < n) out[i] = (_Float16)in[i];
}

__global__ void w_transpose_f16_kernel(const float* __restrict__ in,
                                       _Float16* __restrict__ out) {
  int i = blockIdx.x * blockDim.x + threadIdx.x;
  if (i < DMODEL * DMODEL) {
    int n = i / DMODEL, k = i % DMODEL;
    out[i] = (_Float16)in[(size_t)k * DMODEL + n];
  }
}

// ---------------- LDS-staged double-buffered WMMA GEMM ----------------
// C[M,N] = A[M,K] * B[K,N] + bias. A row-major f16; BT N-major f16.
// Block: 256 threads (8 waves), tile 128M x 128N; wave tile 32M x 64N.
// mode 0: f16 out [B,H,S,HD] (Q,K); mode 1: f16 out [B,H,HD,S] (V^T); mode 2: f32 d_out.
__global__ void __launch_bounds__(256)
gemm_wmma_kernel(const _Float16* __restrict__ A,
                 const _Float16* __restrict__ BT,
                 const float* __restrict__ bias,
                 _Float16* __restrict__ out_h,
                 float* __restrict__ out_f,
                 int mode) {
  __shared__ __align__(16) _Float16 Abuf[2][128 * 32];   // 8 KB each
  __shared__ __align__(16) _Float16 Bbuf[2][128 * 32];

  int tid  = threadIdx.x;
  int lane = tid & 31;
  int w    = tid >> 5;          // 8 waves
  int wm   = w >> 1;            // 0..3 -> M offset wm*32
  int wn   = w & 1;             // 0..1 -> N offset wn*64
  int mblk = blockIdx.y * 128;
  int nblk = blockIdx.x * 128;

  const _Float16* Ag = A  + (size_t)mblk * DMODEL;   // + kb later
  const _Float16* Bg = BT + (size_t)nblk * DMODEL;

  // Each wave issues 4 async b128 instructions per stage (2 for A, 2 for B);
  // whole block fills the 128x32 tiles.
  auto stage_load = [&](int stage, int kb) {
#pragma unroll
    for (int i = 0; i < 2; i++) {
      int u = (w * 2 + i) * 32 + lane;      // 16B unit index, 0..511
      int r = u >> 2, kd = u & 3;           // row, 16B chunk within row
      async_b128(Ag + (size_t)r * DMODEL + kb + kd * 8, &Abuf[stage][u * 8]);
      async_b128(Bg + (size_t)r * DMODEL + kb + kd * 8, &Bbuf[stage][u * 8]);
    }
  };

  v8f acc[2][4] = {};
  stage_load(0, 0);
  int stage = 0;

  for (int kb = 0; kb < DMODEL; kb += 32) {
    bool more = (kb + 32) < DMODEL;
    if (more) {
      stage_load(stage ^ 1, kb + 32);
      asm volatile("s_wait_asynccnt 4" ::: "memory");  // current stage landed
    } else {
      asm volatile("s_wait_asynccnt 0" ::: "memory");
    }
    __syncthreads();

    v16h a0 = load_a_frag(&Abuf[stage][(wm * 32 +  0) * 32], 32, lane);
    v16h a1 = load_a_frag(&Abuf[stage][(wm * 32 + 16) * 32], 32, lane);
#pragma unroll
    for (int t = 0; t < 4; t++) {
      v16h b = load_bt_frag(&Bbuf[stage][(wn * 64 + t * 16) * 32], 32, lane);
      acc[0][t] = wmma_f16(a0, b, acc[0][t]);
      acc[1][t] = wmma_f16(a1, b, acc[1][t]);
    }
    __syncthreads();    // everyone done reading before buffer is re-filled
    stage ^= 1;
  }

  int g = lane >> 4, nl = lane & 15;
#pragma unroll
  for (int i = 0; i < 2; i++) {
#pragma unroll
    for (int t = 0; t < 4; t++) {
#pragma unroll
      for (int r = 0; r < 8; r++) {
        int row = mblk + wm * 32 + i * 16 + g * 8 + r;
        int col = nblk + wn * 64 + t * 16 + nl;
        float v = acc[i][t][r] + bias[col];
        if (mode == 2) {
          out_f[(size_t)row * DMODEL + col] = v;
        } else {
          int b = row >> 11, s = row & (SEQ - 1);
          int h = col >> 7,  d = col & (HDIM - 1);
          size_t o = (mode == 0)
              ? (((size_t)(b * NHEAD + h) * SEQ + s) * HDIM + d)
              : (((size_t)(b * NHEAD + h) * HDIM + d) * SEQ + s);
          out_h[o] = (_Float16)v;
        }
      }
    }
  }
}

// ---------------- flash attention (one wave per 16-query tile per head) ----------------
__global__ void attn_kernel(const _Float16* __restrict__ Qh,   // [B,H,S,HD]
                            const _Float16* __restrict__ Kh,   // [B,H,S,HD]
                            const _Float16* __restrict__ VTh,  // [B,H,HD,S]
                            _Float16* __restrict__ attn_h) {   // [B,S,D]
  __shared__ __align__(32) _Float16 Pls[16 * 32];
  int lane = threadIdx.x & 31;
  int qtile = blockIdx.x * 16;
  int bh = blockIdx.y;
  const _Float16* Q  = Qh  + (size_t)bh * SEQ * HDIM;
  const _Float16* Kp = Kh  + (size_t)bh * SEQ * HDIM;
  const _Float16* VT = VTh + (size_t)bh * HDIM * SEQ;
  int g = lane >> 4, nl = lane & 15;

  v16h qa[4];
#pragma unroll
  for (int c = 0; c < 4; c++)
    qa[c] = load_a_frag(Q + (size_t)qtile * HDIM + c * 32, HDIM, lane);

  v8f acc[8] = {};
  float m_run[8], l_run[8];
#pragma unroll
  for (int r = 0; r < 8; r++) { m_run[r] = -1e30f; l_run[r] = 0.f; }
  const float scale = 0.08838834764831845f;   // 1/sqrt(128)

  // Prime K fragments for jb = 0 (kf[0..3] -> keys jb..jb+15, kf[4..7] -> jb+16..jb+31).
  v16h kf[8];
#pragma unroll
  for (int c = 0; c < 4; c++) {
    kf[c]     = *(const v16h*)(Kp + (size_t)(nl) * HDIM + c * 32 + g * 16);
    kf[4 + c] = *(const v16h*)(Kp + (size_t)(16 + nl) * HDIM + c * 32 + g * 16);
  }

  for (int jb = 0; jb < qtile + 16; jb += 32) {
    // Issue V loads now; first use is after softmax -> overlapped.
    v16h vf[8];
#pragma unroll
    for (int ng = 0; ng < 8; ng++)
      vf[ng] = *(const v16h*)(VT + (size_t)(ng * 16 + nl) * SEQ + jb + g * 16);

    v8f s1 = {}, s2 = {};
#pragma unroll
    for (int c = 0; c < 4; c++) s1 = wmma_f16(qa[c], kf[c], s1);
#pragma unroll
    for (int c = 0; c < 4; c++) s2 = wmma_f16(qa[c], kf[4 + c], s2);

    // Prefetch next block's K fragments (clamped); lands during softmax + PV.
    int jn = (jb + 32 < qtile + 16) ? jb + 32 : jb;
#pragma unroll
    for (int c = 0; c < 4; c++) {
      kf[c]     = *(const v16h*)(Kp + (size_t)(jn + nl) * HDIM + c * 32 + g * 16);
      kf[4 + c] = *(const v16h*)(Kp + (size_t)(jn + 16 + nl) * HDIM + c * 32 + g * 16);
    }

    // Online softmax; row m = g*8+r spans the 16 lanes of this half-wave.
#pragma unroll
    for (int r = 0; r < 8; r++) {
      int q = qtile + g * 8 + r;
      float x1 = s1[r] * scale; if (jb + nl > q)      x1 = -1e30f;
      float x2 = s2[r] * scale; if (jb + 16 + nl > q) x2 = -1e30f;
      float t = fmaxf(x1, x2);
      t = fmaxf(t, __shfl_xor(t, 1, 32));
      t = fmaxf(t, __shfl_xor(t, 2, 32));
      t = fmaxf(t, __shfl_xor(t, 4, 32));
      t = fmaxf(t, __shfl_xor(t, 8, 32));
      float m_new = fmaxf(m_run[r], t);
      float corr = __expf(m_run[r] - m_new);
      float p1 = __expf(x1 - m_new);
      float p2 = __expf(x2 - m_new);
      float rs = p1 + p2;
      rs += __shfl_xor(rs, 1, 32);
      rs += __shfl_xor(rs, 2, 32);
      rs += __shfl_xor(rs, 4, 32);
      rs += __shfl_xor(rs, 8, 32);
      l_run[r] = l_run[r] * corr + rs;
      m_run[r] = m_new;
#pragma unroll
      for (int ng = 0; ng < 8; ng++) acc[ng][r] *= corr;
      int row = g * 8 + r;
      Pls[row * 32 + nl]      = (_Float16)p1;
      Pls[row * 32 + 16 + nl] = (_Float16)p2;
    }
    __syncthreads();

    // Re-layout P (C-layout -> LDS -> A-fragment).
    v16h pa;
    {
      int m = lane & 15;
      v8h lo = *(const v8h*)&Pls[m * 32 + g * 16];
      v8h hi = *(const v8h*)&Pls[m * 32 + g * 16 + 8];
#pragma unroll
      for (int e = 0; e < 8; e++) { pa[e] = lo[e]; pa[8 + e] = hi[e]; }
    }

#pragma unroll
    for (int ng = 0; ng < 8; ng++)
      acc[ng] = wmma_f16(pa, vf[ng], acc[ng]);
    __syncthreads();
  }

  int b = bh >> 4, h = bh & 15;
#pragma unroll
  for (int ng = 0; ng < 8; ng++) {
#pragma unroll
    for (int r = 0; r < 8; r++) {
      int q = qtile + g * 8 + r;
      float v = acc[ng][r] / l_run[r];
      size_t o = ((size_t)(b * SEQ + q)) * DMODEL + h * HDIM + ng * 16 + nl;
      attn_h[o] = (_Float16)v;
    }
  }
}

// ---------------- launch ----------------

extern "C" void kernel_launch(void* const* d_in, const int* in_sizes, int n_in,
                              void* d_out, int out_size, void* d_ws, size_t ws_size,
                              hipStream_t stream) {
  (void)in_sizes; (void)n_in; (void)out_size; (void)ws_size;
  const float* x  = (const float*)d_in[0];
  const float* Wq = (const float*)d_in[1];
  const float* bq = (const float*)d_in[2];
  const float* Wk = (const float*)d_in[3];
  const float* bk = (const float*)d_in[4];
  const float* Wv = (const float*)d_in[5];
  const float* bv = (const float*)d_in[6];
  const float* Wo = (const float*)d_in[7];
  const float* bo = (const float*)d_in[8];
  float* out = (float*)d_out;

  char* ws = (char*)d_ws;
  size_t off = 0;
  auto carve = [&](size_t bytes) -> char* {
    char* p = ws + off;
    off += (bytes + 255) & ~((size_t)255);
    return p;
  };
  const size_t nX = (size_t)MROWS * DMODEL;     // 8M elems
  const size_t nW = (size_t)DMODEL * DMODEL;    // 4M elems
  _Float16* xh  = (_Float16*)carve(nX * 2);
  _Float16* WqT = (_Float16*)carve(nW * 2);
  _Float16* WkT = (_Float16*)carve(nW * 2);
  _Float16* WvT = (_Float16*)carve(nW * 2);
  _Float16* WoT = (_Float16*)carve(nW * 2);
  _Float16* Qh  = (_Float16*)carve(nX * 2);
  _Float16* Kh  = (_Float16*)carve(nX * 2);
  _Float16* VTh = (_Float16*)carve(nX * 2);
  _Float16* attn_h = xh;   // alias: x-f16 dead after V projection

  f32_to_f16_kernel<<<(int)((nX + 255) / 256), 256, 0, stream>>>(x, xh, (int)nX);
  int wblocks = (int)((nW + 255) / 256);
  w_transpose_f16_kernel<<<wblocks, 256, 0, stream>>>(Wq, WqT);
  w_transpose_f16_kernel<<<wblocks, 256, 0, stream>>>(Wk, WkT);
  w_transpose_f16_kernel<<<wblocks, 256, 0, stream>>>(Wv, WvT);
  w_transpose_f16_kernel<<<wblocks, 256, 0, stream>>>(Wo, WoT);

  dim3 ggrid(DMODEL / 128, MROWS / 128);  // (16, 32), 256 threads
  gemm_wmma_kernel<<<ggrid, 256, 0, stream>>>(xh, WqT, bq, Qh,  nullptr, 0);
  gemm_wmma_kernel<<<ggrid, 256, 0, stream>>>(xh, WkT, bk, Kh,  nullptr, 0);
  gemm_wmma_kernel<<<ggrid, 256, 0, stream>>>(xh, WvT, bv, VTh, nullptr, 1);

  dim3 agrid(SEQ / 16, BATCH * NHEAD);    // (128, 32), 1 wave per block
  attn_kernel<<<agrid, 32, 0, stream>>>(Qh, Kh, VTh, attn_h);

  gemm_wmma_kernel<<<ggrid, 256, 0, stream>>>(attn_h, WoT, bo, nullptr, out, 2);
}
